// Attention_30760555774660
// MI455X (gfx1250) — matrix operand order repaired and checked by reading, not compiled, not run on hardware
//
#include <hip/hip_runtime.h>
#include <math.h>

// ---------------------------------------------------------------------------
// CDNA5 (gfx1250) wave32 WMMA bf16 implementation of the stacked-attention net
// ---------------------------------------------------------------------------

typedef __attribute__((ext_vector_type(16))) __bf16 v16bf;
typedef __attribute__((ext_vector_type(8)))  float  v8f;

#define WMMA_BF16(a, b, c) \
  __builtin_amdgcn_wmma_f32_16x16x32_bf16(false, (a), false, (b), (short)0, (c), false, false)

// ---------------------------------------------------------------------------
// A-fragment (16x32 bf16, M x K) for v_wmma_f32_16x16x32_bf16 from row-major
// f32, converting on the fly.  ISA layout (16-bit A 16x32): lanes 0-15 ->
// M=lane, halves 0..7 = K 0..7, halves 8..15 = K 16..23 ; lanes 16-31 ->
// same M, K ranges shifted by +8.
// ---------------------------------------------------------------------------
__device__ __forceinline__ v16bf cvt_a_frag(float4 x0, float4 x1, float4 x2, float4 x3) {
  v16bf a;
  a[0]  = (__bf16)x0.x; a[1]  = (__bf16)x0.y; a[2]  = (__bf16)x0.z; a[3]  = (__bf16)x0.w;
  a[4]  = (__bf16)x1.x; a[5]  = (__bf16)x1.y; a[6]  = (__bf16)x1.z; a[7]  = (__bf16)x1.w;
  a[8]  = (__bf16)x2.x; a[9]  = (__bf16)x2.y; a[10] = (__bf16)x2.z; a[11] = (__bf16)x2.w;
  a[12] = (__bf16)x3.x; a[13] = (__bf16)x3.y; a[14] = (__bf16)x3.z; a[15] = (__bf16)x3.w;
  return a;
}

__device__ __forceinline__ v16bf load_a_frag_f32(const float* __restrict__ A,
                                                 int lda, int m0, int k0) {
  const int lane = threadIdx.x & 31;
  const int m    = lane & 15;
  const int kg   = (lane >> 4) << 3;           // 0 or 8
  const float4* p4 = (const float4*)(A + (size_t)(m0 + m) * lda + (k0 + kg));
  return cvt_a_frag(p4[0], p4[1], p4[4], p4[5]);
}

// A-fragment from an LDS slab stored row-major, 32 f32 per row.
__device__ __forceinline__ v16bf lds_a_frag(const float* s /*shared*/) {
  const int lane = threadIdx.x & 31;
  const int m    = lane & 15;
  const int kg   = (lane >> 4) << 3;
  const float4* p4 = (const float4*)(s + m * 32 + kg);   // ds_load_b128 x4
  return cvt_a_frag(p4[0], p4[1], p4[4], p4[5]);
}

// ---------------------------------------------------------------------------
// Pack a row-major f32 weight [K, N] into bf16 B-fragment order:
//   out[((nt*KT + kt)*32 + lane)*16 + h] =
//       W[(kt*32 + (h>>3)*16 + (lane>>4)*8 + (h&7))*N + nt*16 + (lane&15)]
// Zero-padded for n >= N (final FC, N=3000 -> NT=188).  KT = K/32.
// ---------------------------------------------------------------------------
__global__ void pack_b_kernel(const float* __restrict__ W, __bf16* __restrict__ out,
                              int K, int N, int NT) {
  const int KT = K >> 5;
  size_t total = (size_t)KT * NT * 512;
  size_t idx = (size_t)blockIdx.x * blockDim.x + threadIdx.x;
  if (idx >= total) return;
  int h    = (int)(idx & 15);
  int lane = (int)((idx >> 4) & 31);
  size_t t = idx >> 9;                          // t = nt*KT + kt
  int kt = (int)(t % KT);
  int nt = (int)(t / KT);
  int k  = kt * 32 + ((h & 8) << 1) + ((lane >> 4) << 3) + (h & 7);
  int n  = nt * 16 + (lane & 15);
  float v = (n < N) ? W[(size_t)k * N + n] : 0.0f;
  out[idx] = (__bf16)v;
}

// ---------------------------------------------------------------------------
// Generic GEMM + bias: C[M,N] = A[M,K](f32) * Bpacked(bf16) + bias.
// One wave (32 threads) per 16x16 output tile; grid = (M/16)*NT blocks.
// ---------------------------------------------------------------------------
__global__ void __launch_bounds__(32)
gemm_bias_kernel(const float* __restrict__ A, const __bf16* __restrict__ Bp,
                 const float* __restrict__ bias, float* __restrict__ C,
                 int N, int K, int NT) {
  const int KT   = K >> 5;
  const int nt   = blockIdx.x % NT;
  const int mt   = blockIdx.x / NT;
  const int lane = threadIdx.x & 31;
  const int m0   = mt * 16;
  v8f acc = {};
  for (int kt = 0; kt < KT; ++kt) {
    v16bf a = load_a_frag_f32(A, K, m0, kt * 32);
    v16bf b = *(const v16bf*)(Bp + (((size_t)nt * KT + kt) * 32 + lane) * 16);
    acc = WMMA_BF16(a, b, acc);
  }
  const int n = nt * 16 + (lane & 15);
  if (n < N) {
    const float bv = bias[n];
    const int mh = (lane >> 4) << 3;
#pragma unroll
    for (int r = 0; r < 8; ++r) {
      C[(size_t)(m0 + r + mh) * N + n] = acc[r] + bv;
    }
  }
}

// ---------------------------------------------------------------------------
// Async staging helpers (CDNA5 async-copy to LDS, tracked by ASYNCcnt)
// ---------------------------------------------------------------------------
__device__ __forceinline__ void async_ld_b128(unsigned lds_off, const float* gptr) {
  asm volatile("global_load_async_to_lds_b128 %0, %1, off"
               :: "v"(lds_off), "v"(gptr) : "memory");
}
__device__ __forceinline__ void wait_async_le2() {
  asm volatile("s_wait_asynccnt 0x2" ::: "memory");
}
__device__ __forceinline__ void wait_async_0() {
  asm volatile("s_wait_asynccnt 0x0" ::: "memory");
}

// Issue the 4KB (32 rows x 32 f32) A-slab for K-step kt into LDS buffer.
// 128 threads x 2 b128 chunks.  chunk c: row = c>>3, floats (c&7)*4 .. +3.
__device__ __forceinline__ void issue_slab(const float* __restrict__ img,
                                           int m0, int kt, unsigned lds_base) {
  const int t = threadIdx.x;
#pragma unroll
  for (int i = 0; i < 2; ++i) {
    int c = t + i * 128;
    int r = c >> 3, cg = c & 7;
    const float* g = img + (size_t)(m0 + r) * 1024 + kt * 32 + cg * 4;
    async_ld_b128(lds_base + (unsigned)c * 16u, g);
  }
}

// ---------------------------------------------------------------------------
// Fused attention logits:
//   logits[b,s] = Wp . tanh(img[b,s,:] @ W_ia + Wqavq[b,:]) + bp
// 32 (b,s)-rows per WG (50176/32 = 1568 WGs), 4 waves.  Each wave: 8 N-tiles
// x 2 M-tiles, so every B fragment feeds two WMMAs (halves L2 B-traffic).
// A slab double-buffered in LDS via global_load_async_to_lds_b128.
// ---------------------------------------------------------------------------
__global__ void __launch_bounds__(128)
att_logits_kernel(const float* __restrict__ img,     // [50176,1024]
                  const __bf16* __restrict__ Wia,    // packed, NT=32, KT=32
                  const float* __restrict__ Wq,      // Wqavq [256,512]
                  const float* __restrict__ Wp,      // [512]
                  const float* __restrict__ bp,      // [1]
                  float* __restrict__ logits) {      // [256*196]
  const int KT   = 32;
  const int m0   = blockIdx.x * 32;
  const int wave = threadIdx.x >> 5;
  const int lane = threadIdx.x & 31;
  const int nl   = lane & 15;
  const int hi   = lane >> 4;

  __shared__ float slab[2][32 * 32];   // 2 x 4KB double buffer
  __shared__ float s_logit[32];
  if (threadIdx.x < 32) s_logit[threadIdx.x] = 0.0f;

  const unsigned base0 = (unsigned)(uintptr_t)&slab[0][0];
  const unsigned base1 = (unsigned)(uintptr_t)&slab[1][0];

  issue_slab(img, m0, 0, base0);

  v8f acc0[8] = {};
  v8f acc1[8] = {};

  for (int kt = 0; kt < KT; ++kt) {
    const int cur = kt & 1;
    if (kt + 1 < KT) {
      issue_slab(img, m0, kt + 1, cur ? base0 : base1);
      wait_async_le2();                 // slab for kt has landed (per wave)
    } else {
      wait_async_0();
    }
    __syncthreads();                    // all waves' async writes visible

    v16bf a0 = lds_a_frag(&slab[cur][0]);        // rows m0 .. m0+15
    v16bf a1 = lds_a_frag(&slab[cur][16 * 32]);  // rows m0+16 .. m0+31
#pragma unroll
    for (int j = 0; j < 8; ++j) {
      const int nt = wave * 8 + j;
      v16bf b = *(const v16bf*)(Wia + (((size_t)nt * KT + kt) * 32 + lane) * 16);
      acc0[j] = WMMA_BF16(a0, b, acc0[j]);
      acc1[j] = WMMA_BF16(a1, b, acc1[j]);
    }
    __syncthreads();                    // buffer !cur free for next issue
  }

  // Epilogue: + Wqavq[b,n], tanh, * Wp[n], reduce over n (512 columns).
  float wpj[8];
#pragma unroll
  for (int j = 0; j < 8; ++j) wpj[j] = Wp[(wave * 8 + j) * 16 + nl];

#pragma unroll
  for (int half = 0; half < 2; ++half) {
#pragma unroll
    for (int r = 0; r < 8; ++r) {
      const int m     = half * 16 + r + (hi << 3);
      const int gm    = m0 + m;                  // flat (b*196 + s)
      const int batch = gm / 196;
      const float* wqrow = Wq + (size_t)batch * 512;
      float part = 0.0f;
#pragma unroll
      for (int j = 0; j < 8; ++j) {
        const int n = (wave * 8 + j) * 16 + nl;
        const float v = (half ? acc1[j][r] : acc0[j][r]) + wqrow[n];
        part += tanhf(v) * wpj[j];
      }
      part += __shfl_xor(part, 1, 32);
      part += __shfl_xor(part, 2, 32);
      part += __shfl_xor(part, 4, 32);
      part += __shfl_xor(part, 8, 32);
      if (nl == 0) atomicAdd(&s_logit[m], part);   // ds_add_f32 across waves
    }
  }
  __syncthreads();
  if (threadIdx.x < 32)
    logits[m0 + threadIdx.x] = s_logit[threadIdx.x] + bp[0];
}

// ---------------------------------------------------------------------------
// Softmax over S=196 per batch row.  One 256-thread block per batch.
// ---------------------------------------------------------------------------
__global__ void softmax196_kernel(const float* __restrict__ logits,
                                  float* __restrict__ pi) {
  __shared__ float sd[256];
  const int b = blockIdx.x, t = threadIdx.x;
  float x = (t < 196) ? logits[b * 196 + t] : -1e30f;
  sd[t] = x;
  __syncthreads();
  for (int o = 128; o > 0; o >>= 1) {
    if (t < o) sd[t] = fmaxf(sd[t], sd[t + o]);
    __syncthreads();
  }
  const float mx = sd[0];
  __syncthreads();
  float e = (t < 196) ? __expf(x - mx) : 0.0f;
  sd[t] = e;
  __syncthreads();
  for (int o = 128; o > 0; o >>= 1) {
    if (t < o) sd[t] += sd[t + o];
    __syncthreads();
  }
  const float inv = 1.0f / sd[0];
  if (t < 196) pi[b * 196 + t] = e * inv;
}

// ---------------------------------------------------------------------------
// u[b,d] = sum_s pi[b,s] * img[b,s,d] + qin[b,d].  grid (4, 256), 256 threads.
// ---------------------------------------------------------------------------
__global__ void weighted_sum_kernel(const float* __restrict__ img,
                                    const float* __restrict__ pi,
                                    const float* __restrict__ qin,
                                    float* __restrict__ u) {
  const int b = blockIdx.y;
  const int d = blockIdx.x * 256 + threadIdx.x;
  __shared__ float sp[196];
  if (threadIdx.x < 196) sp[threadIdx.x] = pi[b * 196 + threadIdx.x];
  __syncthreads();
  const float* base = img + (size_t)b * 196 * 1024 + d;
  float acc = 0.0f;
#pragma unroll 4
  for (int s = 0; s < 196; ++s) acc += sp[s] * base[(size_t)s * 1024];
  u[(size_t)b * 1024 + d] = acc + qin[(size_t)b * 1024 + d];
}

// ---------------------------------------------------------------------------
// Host-side orchestration
// ---------------------------------------------------------------------------
extern "C" void kernel_launch(void* const* d_in, const int* in_sizes, int n_in,
                              void* d_out, int out_size, void* d_ws, size_t ws_size,
                              hipStream_t stream) {
  (void)in_sizes; (void)n_in; (void)out_size; (void)ws_size;
  const float* ques  = (const float*)d_in[0];
  const float* img   = (const float*)d_in[1];
  const float* W_ia1 = (const float*)d_in[2];
  const float* W_qa1 = (const float*)d_in[3];
  const float* b_qa1 = (const float*)d_in[4];
  const float* Wp1   = (const float*)d_in[5];
  const float* bp1   = (const float*)d_in[6];
  const float* W_ia2 = (const float*)d_in[7];
  const float* W_qa2 = (const float*)d_in[8];
  const float* b_qa2 = (const float*)d_in[9];
  const float* Wp2   = (const float*)d_in[10];
  const float* bp2   = (const float*)d_in[11];
  const float* W_fc  = (const float*)d_in[12];
  const float* b_fc  = (const float*)d_in[13];
  float* out = (float*)d_out;                   // [256,3000]

  char* ws = (char*)d_ws;
  size_t off = 0;
  auto alloc = [&](size_t bytes) { void* p = ws + off; off = (off + bytes + 255) & ~(size_t)255; return p; };
  const int KT    = 32;    // K = 1024
  const int NT_A  = 32;    // A = 512
  const int NT_FC = 188;   // ceil(3000/16)
  __bf16* Wia1p = (__bf16*)alloc((size_t)KT * NT_A * 512 * 2);
  __bf16* Wqa1p = (__bf16*)alloc((size_t)KT * NT_A * 512 * 2);
  __bf16* Wia2p = (__bf16*)alloc((size_t)KT * NT_A * 512 * 2);
  __bf16* Wqa2p = (__bf16*)alloc((size_t)KT * NT_A * 512 * 2);
  __bf16* Wfcp  = (__bf16*)alloc((size_t)KT * NT_FC * 512 * 2);
  float* wq  = (float*)alloc((size_t)256 * 512 * 4);
  float* lgt = (float*)alloc((size_t)256 * 196 * 4);
  float* pi  = (float*)alloc((size_t)256 * 196 * 4);
  float* u1  = (float*)alloc((size_t)256 * 1024 * 4);
  float* u2  = (float*)alloc((size_t)256 * 1024 * 4);

  {
    size_t tot = (size_t)KT * NT_A * 512;
    int blocks = (int)((tot + 255) / 256);
    pack_b_kernel<<<blocks, 256, 0, stream>>>(W_ia1, Wia1p, 1024, 512, NT_A);
    pack_b_kernel<<<blocks, 256, 0, stream>>>(W_qa1, Wqa1p, 1024, 512, NT_A);
    pack_b_kernel<<<blocks, 256, 0, stream>>>(W_ia2, Wia2p, 1024, 512, NT_A);
    pack_b_kernel<<<blocks, 256, 0, stream>>>(W_qa2, Wqa2p, 1024, 512, NT_A);
    size_t totf = (size_t)KT * NT_FC * 512;
    pack_b_kernel<<<(int)((totf + 255) / 256), 256, 0, stream>>>(W_fc, Wfcp, 1024, 3000, NT_FC);
  }

  const dim3 wsumGrid(4, 256);

  // ---- attention block 1 (query = ques) ----
  gemm_bias_kernel<<<16 * NT_A, 32, 0, stream>>>(ques, Wqa1p, b_qa1, wq, 512, 1024, NT_A);
  att_logits_kernel<<<1568, 128, 0, stream>>>(img, Wia1p, wq, Wp1, bp1, lgt);
  softmax196_kernel<<<256, 256, 0, stream>>>(lgt, pi);
  weighted_sum_kernel<<<wsumGrid, 256, 0, stream>>>(img, pi, ques, u1);

  // ---- attention block 2 (query = u1) ----
  gemm_bias_kernel<<<16 * NT_A, 32, 0, stream>>>(u1, Wqa2p, b_qa2, wq, 512, 1024, NT_A);
  att_logits_kernel<<<1568, 128, 0, stream>>>(img, Wia2p, wq, Wp2, bp2, lgt);
  softmax196_kernel<<<256, 256, 0, stream>>>(lgt, pi);
  weighted_sum_kernel<<<wsumGrid, 256, 0, stream>>>(img, pi, u1, u2);

  // ---- final classifier: out = u2 @ W_fc + b_fc ----
  gemm_bias_kernel<<<16 * NT_FC, 32, 0, stream>>>(u2, Wfcp, b_fc, out, 3000, 1024, NT_FC);
}
